// ScaledDotProductAttention_10462540333414
// MI455X (gfx1250) — matrix operand order repaired and checked by reading, not compiled, not run on hardware
//
#include <hip/hip_runtime.h>
#include <hip/hip_bf16.h>
#include <math.h>

typedef __attribute__((ext_vector_type(16))) __bf16 v16bf;
typedef __attribute__((ext_vector_type(8)))  float  v8f;
typedef __attribute__((ext_vector_type(4)))  unsigned int u32x4;
typedef __attribute__((ext_vector_type(4)))  int i32x4;
typedef __attribute__((ext_vector_type(8)))  int i32x8;

#define BATCH 8
#define SEQ   2048
#define DMODEL 768
#define DHEAD  64
#define SCALE  0.03608439182435161f  /* 1/sqrt(768) */

#if __has_builtin(__builtin_amdgcn_tensor_load_to_lds)
#define HAS_TDM 1
#else
#define HAS_TDM 0
#endif

__device__ __forceinline__ v8f wmma_bf16(v16bf a, v16bf b, v8f c) {
  return __builtin_amdgcn_wmma_f32_16x16x32_bf16(
      /*neg_a=*/false, a, /*neg_b=*/false, b,
      /*c_mod=*/(short)0, c, /*reuse_a=*/false, /*reuse_b=*/false);
}

__device__ __forceinline__ void bf16_split(float f, __bf16& hi, __bf16& lo) {
  hi = (__bf16)f;
  lo = (__bf16)(f - (float)hi);
}

#if HAS_TDM
// byte offset of a __shared__ object within the LDS allocation
__device__ __forceinline__ unsigned lds_addr_of(void* p) {
  return (unsigned)(unsigned long long)(__attribute__((address_space(3))) char*)p;
}

// One-shot TDM copy of 16 KB (8192 bf16 elements) global -> LDS.
// D# built per CDNA5 ISA section 8: 2D tensor, data_size=2B,
// tensor_dim0 = tile_dim0 = 8192, tile_dim1 = 1, type=2 ("image").
__device__ __forceinline__ void tdm_load_16k(const void* gsrc, unsigned lds_off) {
  const unsigned long long ga = (unsigned long long)gsrc;
  const unsigned td0 = 8192;                  // elements
  u32x4 g0;
  g0[0] = 1u;                                  // count=1, user mode
  g0[1] = lds_off;                             // lds_addr (bytes)
  g0[2] = (unsigned)(ga & 0xFFFFFFFFu);        // global_addr[31:0]
  g0[3] = (unsigned)((ga >> 32) & 0x1FFFFFFu)  // global_addr[56:32]
          | (2u << 30);                        // type=2
  i32x8 g1;
  g1[0] = (int)(1u << 16);                     // data_size=1 -> 2 bytes
  g1[1] = (int)((td0 & 0xFFFFu) << 16);        // tensor_dim0[15:0]
  g1[2] = (int)(((td0 >> 16) & 0xFFFFu)        // tensor_dim0[31:16]
          | (1u << 16));                       // tensor_dim1 = 1
  g1[3] = (int)(td0 << 16);                    // tile_dim0 = 8192
  g1[4] = (int)1;                              // tile_dim1 = 1, tile_dim2 = 0
  g1[5] = (int)td0;                            // tensor_dim0_stride[31:0]
  g1[6] = 0;
  g1[7] = 0;
  i32x4 z4 = {};
#if __clang_major__ >= 23
  i32x8 z8 = {};
  __builtin_amdgcn_tensor_load_to_lds(g0, g1, z4, z4, z8, 0);
#else
  __builtin_amdgcn_tensor_load_to_lds(g0, g1, z4, z4, 0);
#endif
}
#endif  // HAS_TDM

// ---------------------------------------------------------------------------
// Kernel 0: pack W[768][64] into WMMA B-fragment order (bf16 hi / lo).
// B order used here (contiguous k, A matches): lane L -> N = L%16 ; element e -> K = (L<16?0:16)+e
// ---------------------------------------------------------------------------
__global__ void pack_w_kernel(const float* __restrict__ qw,
                              const float* __restrict__ kw,
                              const float* __restrict__ vw,
                              __bf16* __restrict__ whi,
                              __bf16* __restrict__ wlo) {
  const int lane  = threadIdx.x;      // 32 threads / block
  const int frag  = blockIdx.x;       // 0..95 : chunk*4 + ntile
  const int mat   = blockIdx.y;       // 0..2
  const int chunk = frag >> 2;
  const int ntile = frag & 3;
  const float* W = (mat == 0) ? qw : (mat == 1) ? kw : vw;
  const int n     = ntile * 16 + (lane & 15);
  const int kbase = chunk * 32 + ((lane < 16) ? 0 : 16);
  const size_t out = ((size_t)(mat * 96 + frag)) * 512 + (size_t)lane * 16;
  #pragma unroll
  for (int e = 0; e < 16; ++e) {
    float w = W[(size_t)(kbase + e) * DHEAD + n];
    __bf16 hi, lo; bf16_split(w, hi, lo);
    whi[out + e] = hi;
    wlo[out + e] = lo;
  }
}

// ---------------------------------------------------------------------------
// Kernel 1: QKV projection.  [B*S,768] x [768,64] (+bias) -> Q,K,V fp32.
// ---------------------------------------------------------------------------
__global__ void __launch_bounds__(256) qkv_kernel(
    const float* __restrict__ x,
    const float* __restrict__ qb, const float* __restrict__ kb,
    const float* __restrict__ vb,
    const __bf16* __restrict__ whi, const __bf16* __restrict__ wlo,
    float* __restrict__ Q, float* __restrict__ K, float* __restrict__ V) {
  const int lane    = threadIdx.x & 31;
  const int wave    = threadIdx.x >> 5;
  const int rowbase = blockIdx.x * 128 + wave * 16;
  const int lrow    = lane & 15;
  const bool hi16   = lane >= 16;
  const int arow    = rowbase + lrow;

  v8f acc[3][4];
  #pragma unroll
  for (int m = 0; m < 3; ++m)
    #pragma unroll
    for (int t = 0; t < 4; ++t) acc[m][t] = v8f{};

  for (int chunk = 0; chunk < 24; ++chunk) {
    const float* xp = x + (size_t)arow * DMODEL + chunk * 32 + (hi16 ? 16 : 0);
    v16bf ahi, alo;
    #pragma unroll
    for (int e = 0; e < 8; ++e) {
      __bf16 h, l; bf16_split(xp[e], h, l);
      ahi[e] = h; alo[e] = l;
    }
    #pragma unroll
    for (int e = 8; e < 16; ++e) {           // K = base + e
      __bf16 h, l; bf16_split(xp[e], h, l);
      ahi[e] = h; alo[e] = l;
    }
    #pragma unroll
    for (int m = 0; m < 3; ++m) {
      #pragma unroll
      for (int t = 0; t < 4; ++t) {
        const size_t fi =
            ((size_t)(m * 96 + chunk * 4 + t)) * 512 + (size_t)lane * 16;
        v16bf bhi = *(const v16bf*)(whi + fi);
        v16bf blo = *(const v16bf*)(wlo + fi);
        v8f c = acc[m][t];
        c = wmma_bf16(ahi, bhi, c);
        c = wmma_bf16(ahi, blo, c);
        c = wmma_bf16(alo, bhi, c);
        acc[m][t] = c;
      }
    }
  }

  float* outs[3] = {Q, K, V};
  const float* biases[3] = {qb, kb, vb};
  #pragma unroll
  for (int m = 0; m < 3; ++m) {
    #pragma unroll
    for (int t = 0; t < 4; ++t) {
      const int h = t * 16 + lrow;
      const float bias = biases[m][h];
      #pragma unroll
      for (int r = 0; r < 8; ++r) {
        const int orow = rowbase + r + (hi16 ? 8 : 0);
        outs[m][(size_t)orow * DHEAD + h] = acc[m][t][r] + bias;
      }
    }
  }
}

// ---------------------------------------------------------------------------
// Kernel 2: pack K and V (fp32) into per-chunk WMMA B-fragment blocks.
// Per (b, 32-key chunk c): 16 fragments x 512 bf16 = 16 KB, contiguous:
//   f = kt*4 + hc*2 + p          (K fragments, kt in 0..1, hc in 0..1, p=hi/lo)
//   f = 8 + t*2 + p              (V fragments, t head tile 0..3)
// ---------------------------------------------------------------------------
__global__ void __launch_bounds__(256) pack_kv_kernel(
    const float* __restrict__ Kf, const float* __restrict__ Vf,
    __bf16* __restrict__ kv) {
  const int lane = threadIdx.x & 31;
  const int wave = threadIdx.x >> 5;
  const int c = blockIdx.x;            // 0..63
  const int b = blockIdx.y;            // 0..7
  const int lrow = lane & 15;
  const bool hi16 = lane >= 16;
  const size_t base = ((size_t)(b * 64 + c) * 16) * 512 + (size_t)lane * 16;

  if (wave < 4) {                      // K fragments: B layout for Q*K^T
    const int kt = wave >> 1, hc = wave & 1;
    const int key = c * 32 + kt * 16 + lrow;
    const float* src =
        Kf + ((size_t)b * SEQ + key) * DHEAD + hc * 32 + (hi16 ? 16 : 0);
    const size_t fo = base + (size_t)(kt * 4 + hc * 2) * 512;
    #pragma unroll
    for (int e = 0; e < 16; ++e) {
      __bf16 h, l; bf16_split(src[e], h, l);
      kv[fo + e] = h;
      kv[fo + 512 + e] = l;
    }
  } else {                             // V fragments: B layout for P*V
    const int t = wave - 4;
    const int h = t * 16 + lrow;
    const float* src =
        Vf + ((size_t)b * SEQ + c * 32 + (hi16 ? 16 : 0)) * DHEAD + h;
    const size_t fo = base + (size_t)(8 + t * 2) * 512;
    #pragma unroll
    for (int e = 0; e < 16; ++e) {
      __bf16 hh, ll; bf16_split(src[(size_t)e * DHEAD], hh, ll);
      kv[fo + e] = hh;
      kv[fo + 512 + e] = ll;
    }
  }
}

// ---------------------------------------------------------------------------
// Kernel 3: streaming (flash) attention.
// Per block: 128 query rows of one batch (8 waves x 16-query tiles).
// Per 32-key chunk: TDM stages the 16 KB K/V fragment block into LDS once,
// all waves consume it: 12 score WMMAs + online softmax + 8 P*V WMMAs.
// ---------------------------------------------------------------------------
__global__ void __launch_bounds__(256) attn_kernel(
    const float* __restrict__ Q, const __bf16* __restrict__ KV,
    const unsigned char* __restrict__ mask, float* __restrict__ out) {
  const int lane  = threadIdx.x & 31;
  const int wave  = threadIdx.x >> 5;
  const int b     = blockIdx.y;
  const int qbase = blockIdx.x * 128 + wave * 16;
  const int lrow  = lane & 15;
  const bool hi16 = lane >= 16;
  const int abase = hi16 ? 8 : 0;

  __shared__ __align__(64) __bf16 kvbuf[16 * 512];  // 16 KB staged K/V frags
  __shared__ __align__(64) __bf16 pshare[8][16][32]; // per-wave P transpose

  // ---- Q tile as A fragments (two 32-wide head chunks, hi/lo) ----
  const float* qrow = Q + ((size_t)b * SEQ + qbase + lrow) * DHEAD;
  v16bf qh[2], ql[2];
  #pragma unroll
  for (int c2 = 0; c2 < 2; ++c2) {
    const float* qp = qrow + c2 * 32 + (hi16 ? 16 : 0);
    #pragma unroll
    for (int e = 0; e < 8; ++e) {
      __bf16 h, l; bf16_split(qp[e], h, l);
      qh[c2][e] = h; ql[c2][e] = l;
    }
    #pragma unroll
    for (int e = 8; e < 16; ++e) {
      __bf16 h, l; bf16_split(qp[e], h, l);
      qh[c2][e] = h; ql[c2][e] = l;
    }
  }

  float mrow[8], lsum[8];
  v8f oacc[4];
  #pragma unroll
  for (int r = 0; r < 8; ++r) { mrow[r] = -INFINITY; lsum[r] = 0.0f; }
  #pragma unroll
  for (int t = 0; t < 4; ++t) oacc[t] = v8f{};

#if HAS_TDM
  const unsigned kv_lds = lds_addr_of(kvbuf);
#endif

  for (int c = 0; c < 64; ++c) {          // 64 chunks of 32 keys
    const __bf16* kvg = KV + ((size_t)(b * 64 + c) * 16) * 512;

#if HAS_TDM
    if (wave == 0) {
      tdm_load_16k(kvg, kv_lds);
      __builtin_amdgcn_s_wait_tensorcnt(0);
    }
    if (wave == 1 && c + 1 < 64) {        // prefetch next fragment block
      __builtin_prefetch(kvg + 16 * 512 + lane * 512, 0, 1);
    }
#else
    for (int u = threadIdx.x; u < 16 * 512; u += 256) kvbuf[u] = kvg[u];
#endif
    __syncthreads();

    float sc[2][8];
    float cmax[8];
    #pragma unroll
    for (int r = 0; r < 8; ++r) cmax[r] = -INFINITY;

    #pragma unroll
    for (int kt = 0; kt < 2; ++kt) {      // two 16-key score tiles
      const __bf16* kf = kvbuf + (size_t)(kt * 4) * 512 + lane * 16;
      v16bf kh0 = *(const v16bf*)(kf);
      v16bf kl0 = *(const v16bf*)(kf + 512);
      v16bf kh1 = *(const v16bf*)(kf + 1024);
      v16bf kl1 = *(const v16bf*)(kf + 1536);

      v8f s = v8f{};
      s = wmma_bf16(qh[0], kh0, s);
      s = wmma_bf16(qh[0], kl0, s);
      s = wmma_bf16(ql[0], kh0, s);
      s = wmma_bf16(qh[1], kh1, s);
      s = wmma_bf16(qh[1], kl1, s);
      s = wmma_bf16(ql[1], kh1, s);

      const int key = c * 32 + kt * 16 + lrow;
      const unsigned char mk = mask[(size_t)b * SEQ + key];
      #pragma unroll
      for (int r = 0; r < 8; ++r) {
        const float v = mk ? s[r] * SCALE : -INFINITY;
        sc[kt][r] = v;
        cmax[r] = fmaxf(cmax[r], v);
      }
    }

    // row-max across the 16-lane column group (wave32 shfl)
    #pragma unroll
    for (int r = 0; r < 8; ++r) {
      #pragma unroll
      for (int m = 8; m >= 1; m >>= 1)
        cmax[r] = fmaxf(cmax[r], __shfl_xor(cmax[r], m, 32));
    }

    // online-softmax rescale
    float pfac[8];
    #pragma unroll
    for (int r = 0; r < 8; ++r) {
      const float mnew = fmaxf(mrow[r], cmax[r]);
      float corr = __expf(mrow[r] - mnew);
      if (mnew == -INFINITY) corr = 1.0f;
      mrow[r] = mnew;
      pfac[r] = corr;
      lsum[r] *= corr;
    }
    #pragma unroll
    for (int t = 0; t < 4; ++t)
      #pragma unroll
      for (int r = 0; r < 8; ++r) oacc[t][r] *= pfac[r];

    // P = exp(s - m); stage into LDS in (row, key) order
    float psum[8];
    #pragma unroll
    for (int r = 0; r < 8; ++r) psum[r] = 0.0f;
    #pragma unroll
    for (int kt = 0; kt < 2; ++kt) {
      #pragma unroll
      for (int r = 0; r < 8; ++r) {
        float p = __expf(sc[kt][r] - mrow[r]);
        if (mrow[r] == -INFINITY) p = 0.0f;
        psum[r] += p;
        pshare[wave][r + abase][kt * 16 + lrow] = (__bf16)p;
      }
    }
    #pragma unroll
    for (int r = 0; r < 8; ++r) {
      #pragma unroll
      for (int m = 8; m >= 1; m >>= 1)
        psum[r] += __shfl_xor(psum[r], m, 32);
      lsum[r] += psum[r];
    }
    asm volatile("" ::: "memory");   // keep LDS writes before reads

    // reload P as A fragment (transpose through LDS)
    v16bf pa;
    const __bf16* pp = &pshare[wave][lrow][hi16 ? 16 : 0];
    #pragma unroll
    for (int e = 0; e < 8; ++e)  pa[e] = pp[e];
    #pragma unroll
    for (int e = 8; e < 16; ++e) pa[e] = pp[e];   // K = base + e

    // P x V : 4 head tiles, V hi/lo from staged fragments
    #pragma unroll
    for (int t = 0; t < 4; ++t) {
      const __bf16* vf = kvbuf + (size_t)(8 + t * 2) * 512 + lane * 16;
      v16bf vh = *(const v16bf*)(vf);
      v16bf vl = *(const v16bf*)(vf + 512);
      oacc[t] = wmma_bf16(pa, vh, oacc[t]);
      oacc[t] = wmma_bf16(pa, vl, oacc[t]);
    }
    __syncthreads();                 // protect kvbuf before next TDM load
  }

  // epilogue: out = acc / l
  #pragma unroll
  for (int t = 0; t < 4; ++t) {
    const int h = t * 16 + lrow;
    #pragma unroll
    for (int r = 0; r < 8; ++r) {
      const int qr = qbase + r + abase;
      out[((size_t)b * SEQ + qr) * DHEAD + h] = oacc[t][r] / lsum[r];
    }
  }
}

// ---------------------------------------------------------------------------
extern "C" void kernel_launch(void* const* d_in, const int* in_sizes, int n_in,
                              void* d_out, int out_size, void* d_ws,
                              size_t ws_size, hipStream_t stream) {
  const float* x  = (const float*)d_in[0];
  const unsigned char* amask = (const unsigned char*)d_in[1];
  const float* qw = (const float*)d_in[2];
  const float* qb = (const float*)d_in[3];
  const float* kw = (const float*)d_in[4];
  const float* kb = (const float*)d_in[5];
  const float* vw = (const float*)d_in[6];
  const float* vb = (const float*)d_in[7];
  float* out = (float*)d_out;

  // workspace layout
  const size_t nQKV = (size_t)BATCH * SEQ * DHEAD;     // 1 Mi floats each
  float* Q = (float*)d_ws;                             // 4 MB
  float* K = Q + nQKV;                                 // 4 MB
  float* V = K + nQKV;                                 // 4 MB
  __bf16* whi = (__bf16*)(V + nQKV);                   // 288 KB
  __bf16* wlo = whi + (size_t)3 * 96 * 512;            // 288 KB
  __bf16* KVfrag = wlo + (size_t)3 * 96 * 512;         // 8 MB

  pack_w_kernel<<<dim3(96, 3), 32, 0, stream>>>(qw, kw, vw, whi, wlo);
  qkv_kernel<<<dim3(BATCH * SEQ / 128), 256, 0, stream>>>(
      x, qb, kb, vb, whi, wlo, Q, K, V);
  pack_kv_kernel<<<dim3(64, BATCH), 256, 0, stream>>>(K, V, KVfrag);
  attn_kernel<<<dim3(SEQ / 128, BATCH), 256, 0, stream>>>(Q, KVfrag, amask, out);
}
